// GAT_85676007620917
// MI455X (gfx1250) — matrix operand order repaired and checked by reading, not compiled, not run on hardware
//
#include <hip/hip_runtime.h>
#include <hip/hip_bf16.h>

// GAT x4 + (h1 h1^T) h4 + row-normalize, restructured for MI455X (gfx1250):
//  - all big GEMMs via v_wmma_f32_16x16x32_bf16 (wave32 fragments)
//  - attention N x N matrix never materialized (flash-style 2-pass softmax)
//  - s = h1@h1^T eliminated via associativity: z1 = h1 @ (h1^T @ h4)
//  - attention B-operand tiles staged to LDS by the Tensor Data Mover
//    (tensor_load_to_lds + s_wait_tensorcnt), double-buffered one slab ahead;
//    guarded fallback = synchronous global->LDS staging.

typedef __attribute__((ext_vector_type(16))) __bf16 v16bf;
typedef __attribute__((ext_vector_type(8)))  __bf16 v8bf;
typedef __attribute__((ext_vector_type(8)))  float  v8f;
typedef __attribute__((ext_vector_type(4)))  unsigned int u32x4;
typedef __attribute__((ext_vector_type(8)))  int i32x8;
typedef __attribute__((ext_vector_type(4)))  int i32x4;

#define NN 4096
static constexpr float kAlpha  = 0.2f;     // LeakyReLU slope
static constexpr float kSigma  = 0.4f;     // skip mixing
static constexpr float kNegBig = -9.0e15f;

#if defined(__HIP_DEVICE_COMPILE__) && defined(__has_builtin)
#if __has_builtin(__builtin_amdgcn_tensor_load_to_lds) && \
    __has_builtin(__builtin_amdgcn_s_wait_tensorcnt)
#define USE_TDM 1
#endif
#endif

__device__ __forceinline__ unsigned short f32_bf16(float f) {
  unsigned int u = __float_as_uint(f);
  u += 0x7fffu + ((u >> 16) & 1u);         // round-to-nearest-even
  return (unsigned short)(u >> 16);
}

#ifdef USE_TDM
// 2D TDM load: tile of tileRows rows x tileK bf16 elements from a row-major
// tensor (row stride = strideElems bf16) into contiguous LDS rows.
__device__ __forceinline__ void tdm_load_2d(const unsigned short* gsrc,
                                            unsigned ldsByteOff,
                                            int tileK, int tileRows,
                                            int strideElems, int tensorRows) {
  unsigned long long va = (unsigned long long)gsrc;
  u32x4 g0;
  g0[0] = 1u;                                                // count=1, load
  g0[1] = ldsByteOff;                                        // lds_addr
  g0[2] = (unsigned)va;                                      // global_addr lo
  g0[3] = ((unsigned)(va >> 32) & 0x01FFFFFFu) | 0x80000000u; // hi | type=2
  i32x8 g1;
  g1[0] = (int)(1u << 16);                                   // data_size=2B
  g1[1] = (int)(((unsigned)strideElems & 0xFFFFu) << 16);    // tensor_dim0 lo16
  g1[2] = (int)((((unsigned)strideElems >> 16) & 0xFFFFu) |
                (((unsigned)tensorRows & 0xFFFFu) << 16));   // dim0 hi | dim1 lo
  g1[3] = (int)((((unsigned)tensorRows >> 16) & 0xFFFFu) |
                (((unsigned)tileK & 0xFFFFu) << 16));        // dim1 hi | tile_dim0
  g1[4] = (int)((unsigned)tileRows & 0xFFFFu);               // tile_dim1, tile_dim2=0
  g1[5] = strideElems;                                       // dim0_stride lo32
  g1[6] = 0;
  g1[7] = 0;
  i32x4 z4 = {};
#if __clang_major__ >= 23
  i32x8 z8 = {};
  __builtin_amdgcn_tensor_load_to_lds(g0, g1, z4, z4, z8, 0);
#else
  __builtin_amdgcn_tensor_load_to_lds(g0, g1, z4, z4, 0);
#endif
}
#endif

// ---------------- elementwise converts ----------------
__global__ void k_convert_bf16(const float* __restrict__ src,
                               unsigned short* __restrict__ dst, int count) {
  int i = blockIdx.x * blockDim.x + threadIdx.x;
  if (i < count) dst[i] = f32_bf16(src[i]);
}

__global__ void k_mix_bf16(const float* __restrict__ prev,
                           const float* __restrict__ skip,
                           unsigned short* __restrict__ dst, int count) {
  int i = blockIdx.x * blockDim.x + threadIdx.x;
  if (i < count) dst[i] = f32_bf16((1.0f - kSigma) * prev[i] + kSigma * skip[i]);
}

// src [R,C] f32  ->  dst [C,R] bf16   (LDS-tiled transpose)
__global__ void k_transpose_bf16(const float* __restrict__ src,
                                 unsigned short* __restrict__ dst, int R, int C) {
  __shared__ unsigned short tile[32][33];
  int c0 = blockIdx.x * 32, r0 = blockIdx.y * 32;
  for (int rr = threadIdx.y; rr < 32; rr += 8) {
    int r = r0 + rr, c = c0 + threadIdx.x;
    float v = (r < R && c < C) ? src[(size_t)r * C + c] : 0.0f;
    tile[rr][threadIdx.x] = f32_bf16(v);
  }
  __syncthreads();
  for (int cc = threadIdx.y; cc < 32; cc += 8) {
    int c = c0 + cc, r = r0 + threadIdx.x;
    if (c < C && r < R) dst[(size_t)c * R + r] = tile[threadIdx.x][cc];
  }
}

// ---------------- generic WMMA GEMM ----------------
// C[M,N] f32 = A[M,K] bf16(row-major) x B, with B supplied transposed: BT[N,K].
// One wave = one 16x16 tile. grid = (N/16, M/16).
__global__ void __launch_bounds__(32)
k_wmma_gemm(const unsigned short* __restrict__ A,
            const unsigned short* __restrict__ BT,
            float* __restrict__ C, int Ndim, int Kdim) {
  const int lane = threadIdx.x;
  const int mrow = blockIdx.y * 16 + (lane & 15);
  const int nrow = blockIdx.x * 16 + (lane & 15);
  const int half = lane >> 4;
  const unsigned short* arow = A  + (size_t)mrow * Kdim;
  const unsigned short* brow = BT + (size_t)nrow * Kdim;
  v8f acc = {};
  for (int kb = 0; kb < Kdim; kb += 32) {
    v8bf a0 = *reinterpret_cast<const v8bf*>(arow + kb + half * 16);
    v8bf a1 = *reinterpret_cast<const v8bf*>(arow + kb + half * 16 + 8);
    v16bf av = __builtin_shufflevector(a0, a1, 0,1,2,3,4,5,6,7,8,9,10,11,12,13,14,15);
    v16bf bv = *reinterpret_cast<const v16bf*>(brow + kb + half * 16);
    acc = __builtin_amdgcn_wmma_f32_16x16x32_bf16(false, av, false, bv,
                                                  (short)0, acc, false, false);
  }
  const int ccol  = blockIdx.x * 16 + (lane & 15);
  const int rbase = blockIdx.y * 16 + 8 * half;
#pragma unroll
  for (int r = 0; r < 8; ++r)
    C[(size_t)(rbase + r) * Ndim + ccol] = acc[r];
}

// ---------------- attention scores: es = h a_self, en = h a_neigh ----------------
__global__ void __launch_bounds__(32)
k_gemv2(const float* __restrict__ h, const float* __restrict__ a_s,
        const float* __restrict__ a_n, float* __restrict__ es,
        float* __restrict__ en, int d) {
  const int row = blockIdx.x, lane = threadIdx.x;
  const float* hr = h + (size_t)row * d;
  float s0 = 0.f, s1 = 0.f;
  for (int j = lane; j < d; j += 32) {
    float v = hr[j];
    s0 += v * a_s[j];
    s1 += v * a_n[j];
  }
#pragma unroll
  for (int o = 16; o > 0; o >>= 1) {
    s0 += __shfl_xor(s0, o, 32);
    s1 += __shfl_xor(s1, o, 32);
  }
  if (lane == 0) { es[row] = s0; en[row] = s1; }
}

// ---------------- pass A: per-row online softmax stats (max, 1/sum) ----------------
__global__ void __launch_bounds__(32)
k_stats(const float* __restrict__ es, const float* __restrict__ en,
        const float* __restrict__ Mmat, const float* __restrict__ adj,
        float* __restrict__ mrow, float* __restrict__ rs) {
  const int i = blockIdx.x, lane = threadIdx.x;
  const float esi = es[i];
  const float* Mr = Mmat + (size_t)i * NN;
  const float* Ar = adj  + (size_t)i * NN;
  float m = -3.0e38f, s = 0.0f;
  for (int j = lane; j < NN; j += 32) {
    __builtin_prefetch(Mr + j + 512, 0, 0);
    __builtin_prefetch(Ar + j + 512, 0, 0);
    float e = (esi + en[j]) * Mr[j];
    e = e > 0.f ? e : kAlpha * e;
    e = Ar[j] > 0.f ? e : kNegBig;
    if (e > m) { s = s * __expf(m - e) + 1.0f; m = e; }
    else       { s += __expf(e - m); }
  }
#pragma unroll
  for (int o = 16; o > 0; o >>= 1) {
    float mo = __shfl_xor(m, o, 32), so = __shfl_xor(s, o, 32);
    float mn = fmaxf(m, mo);
    s = s * __expf(m - mn) + so * __expf(mo - mn);
    m = mn;
  }
  if (lane == 0) { mrow[i] = m; rs[i] = 1.0f / s; }
}

// ---------------- pass B: out = elu(P @ h), P generated on the fly ----------------
// Block = 4 waves = 16 rows x 64 cols of output. k-slab = 64, double-buffered:
//   A operand (P tile, bf16, exact WMMA fragment layout) built in LDS by VALU;
//   B operand (hT tile [64 cols][64 k]) DMA'd to LDS by the TDM one slab ahead.
// One workgroup barrier per slab, two WMMAs per slab.
__global__ void __launch_bounds__(128)
k_attn_gemm(const float* __restrict__ Mmat, const float* __restrict__ adj,
            const float* __restrict__ es, const float* __restrict__ en,
            const float* __restrict__ mrow, const float* __restrict__ rs,
            const unsigned short* __restrict__ hT, float* __restrict__ out, int d) {
  __shared__ __align__(32) unsigned short Abuf[2][2][32][16]; // [par][sub][lane][elem]
  __shared__ __align__(16) unsigned short Bbuf[2][64][64];    // [par][col][k]
  const int tid  = threadIdx.x;
  const int wave = tid >> 5;
  const int lane = tid & 31;
  const int i0   = blockIdx.x * 16;
  const int colTile0 = blockIdx.y * 64;
  const int colBase  = colTile0 + wave * 16;
  const bool active  = colBase < d;
  const int nrel  = wave * 16 + (lane & 15);   // column within block tile
  const int halfk = lane >> 4;

  // P generation: this thread covers k-offset pj of each 32-wide subtile,
  // rows pm, pm+4, pm+8, pm+12 of the 16-row block.
  const int pj = tid & 31;
  const int pm = tid >> 5;
  const int alane = 16 * (pj >> 4);                    // + m
  const int aelem = pj & 15;
  float esv[4], mv[4], rv[4];
  const float* Mr[4];
  const float* Ar[4];
#pragma unroll
  for (int q = 0; q < 4; ++q) {
    const int row = i0 + pm + 4 * q;
    esv[q] = es[row]; mv[q] = mrow[row]; rv[q] = rs[row];
    Mr[q] = Mmat + (size_t)row * NN;
    Ar[q] = adj  + (size_t)row * NN;
  }

#ifdef USE_TDM
  const unsigned ldsB0 = (unsigned)(unsigned long long)(const void*)&Bbuf[0][0][0];
  // prologue: DMA slab 0 into buffer 0
  if (wave == 0)
    tdm_load_2d(hT + (size_t)colTile0 * NN, ldsB0, 64, 64, NN, 512);
#else
  {
    // synchronous staging fallback for slab 0
#pragma unroll
    for (int r = 0; r < 4; ++r) {
      const int c = tid + 128 * r;
      const int row = c >> 3, koff = (c & 7) * 8;
      uint4 v = *reinterpret_cast<const uint4*>(
          hT + (size_t)(colTile0 + row) * NN + koff);
      *reinterpret_cast<uint4*>(&Bbuf[0][row][koff]) = v;
    }
  }
#endif

  v8f acc = {};
  for (int it = 0; it < NN / 64; ++it) {
    const int kb  = it * 64;
    const int par = it & 1;
    // ---- build P tile for this slab (both 32-wide subtiles) ----
#pragma unroll
    for (int sub = 0; sub < 2; ++sub) {
      const int j = kb + sub * 32 + pj;
      const float enj = en[j];
#pragma unroll
      for (int q = 0; q < 4; ++q) {
        float e = (esv[q] + enj) * Mr[q][j];
        e = e > 0.f ? e : kAlpha * e;
        float p = (Ar[q][j] > 0.f) ? __expf(e - mv[q]) * rv[q] : 0.0f;
        Abuf[par][sub][pm + 4 * q + alane][aelem] = f32_bf16(p);
      }
    }
    // ---- stage next B slab into the other buffer ----
    const bool hasNext = (kb + 64) < NN;
#ifdef USE_TDM
    if (wave == 0) {
      if (hasNext)
        tdm_load_2d(hT + (size_t)colTile0 * NN + kb + 64,
                    ldsB0 + (unsigned)(par ^ 1) * sizeof(Bbuf[0]),
                    64, 64, NN, 512);
      if (hasNext) __builtin_amdgcn_s_wait_tensorcnt(1);
      else         __builtin_amdgcn_s_wait_tensorcnt(0);
    }
#else
    if (hasNext) {
#pragma unroll
      for (int r = 0; r < 4; ++r) {
        const int c = tid + 128 * r;
        const int row = c >> 3, koff = (c & 7) * 8;
        uint4 v = *reinterpret_cast<const uint4*>(
            hT + (size_t)(colTile0 + row) * NN + kb + 64 + koff);
        *reinterpret_cast<uint4*>(&Bbuf[par ^ 1][row][koff]) = v;
      }
    }
#endif
    __syncthreads();   // P visible, current B slab landed, prior reads done
    // ---- two WMMAs on the slab ----
#pragma unroll
    for (int sub = 0; sub < 2; ++sub) {
      v16bf av = *reinterpret_cast<const v16bf*>(&Abuf[par][sub][lane][0]);
      v16bf bv = *reinterpret_cast<const v16bf*>(&Bbuf[par][nrel][sub * 32 + halfk * 16]);
      acc = __builtin_amdgcn_wmma_f32_16x16x32_bf16(false, av, false, bv,
                                                    (short)0, acc, false, false);
    }
  }
  if (active) {
    const int ccol  = colBase + (lane & 15);
    const int rbase = i0 + 8 * halfk;
#pragma unroll
    for (int r = 0; r < 8; ++r) {
      float x = acc[r];
      out[(size_t)(rbase + r) * d + ccol] = x > 0.f ? x : __expf(x) - 1.0f;  // ELU
    }
  }
}

// ---------------- tail: tmp = h1^T @ h4  [512,16] ----------------
__global__ void __launch_bounds__(32)
k_h1t_h4(const float* __restrict__ h1, const float* __restrict__ h4,
         float* __restrict__ tmp) {
  const int m = blockIdx.x, lane = threadIdx.x;
  float acc[16];
#pragma unroll
  for (int n = 0; n < 16; ++n) acc[n] = 0.f;
  for (int i = lane; i < NN; i += 32) {
    const float a = h1[(size_t)i * 512 + m];
    const float* hr = h4 + (size_t)i * 16;
#pragma unroll
    for (int n = 0; n < 16; ++n) acc[n] += a * hr[n];
  }
#pragma unroll
  for (int n = 0; n < 16; ++n)
#pragma unroll
    for (int o = 16; o > 0; o >>= 1) acc[n] += __shfl_xor(acc[n], o, 32);
  if (lane == 0)
#pragma unroll
    for (int n = 0; n < 16; ++n) tmp[m * 16 + n] = acc[n];
}

// ---------------- tail: z = normalize(h1 @ tmp) ----------------
__global__ void __launch_bounds__(32)
k_z(const float* __restrict__ h1, const float* __restrict__ tmp,
    float* __restrict__ z) {
  const int i = blockIdx.x, lane = threadIdx.x;
  const float* hr = h1 + (size_t)i * 512;
  float acc[16];
#pragma unroll
  for (int n = 0; n < 16; ++n) acc[n] = 0.f;
  for (int m = lane; m < 512; m += 32) {
    const float a = hr[m];
    const float* tr = tmp + m * 16;
#pragma unroll
    for (int n = 0; n < 16; ++n) acc[n] += a * tr[n];
  }
#pragma unroll
  for (int n = 0; n < 16; ++n)
#pragma unroll
    for (int o = 16; o > 0; o >>= 1) acc[n] += __shfl_xor(acc[n], o, 32);
  if (lane == 0) {
    float s = 0.f;
#pragma unroll
    for (int n = 0; n < 16; ++n) s += acc[n] * acc[n];
    const float inv = 1.0f / fmaxf(sqrtf(s), 1e-12f);
#pragma unroll
    for (int n = 0; n < 16; ++n) z[(size_t)i * 16 + n] = acc[n] * inv;
  }
}

// ---------------- host orchestration ----------------
extern "C" void kernel_launch(void* const* d_in, const int* in_sizes, int n_in,
                              void* d_out, int out_size, void* d_ws, size_t ws_size,
                              hipStream_t stream) {
  (void)in_sizes; (void)n_in; (void)out_size; (void)ws_size;
  const float* x    = (const float*)d_in[0];
  const float* adj  = (const float*)d_in[1];
  const float* Mmat = (const float*)d_in[2];
  const float* skips[4] = {nullptr, (const float*)d_in[3],
                           (const float*)d_in[4], (const float*)d_in[5]};
  const float* W[4]  = {(const float*)d_in[6],  (const float*)d_in[9],
                        (const float*)d_in[12], (const float*)d_in[15]};
  const float* As[4] = {(const float*)d_in[7],  (const float*)d_in[10],
                        (const float*)d_in[13], (const float*)d_in[16]};
  const float* An[4] = {(const float*)d_in[8],  (const float*)d_in[11],
                        (const float*)d_in[14], (const float*)d_in[17]};
  const int Kin[4]  = {1024, 512, 256, 64};
  const int dOut[4] = {512, 256, 64, 16};

  char* ws = (char*)d_ws;
  unsigned short* mixB = (unsigned short*)(ws + 0);          // [4096,1024] bf16
  unsigned short* WT   = (unsigned short*)(ws + 8388608);    // [d,K] bf16
  float*          hW   = (float*)         (ws + 9437184);    // [4096,512] f32
  unsigned short* hWT  = (unsigned short*)(ws + 17825792);   // [512,4096] bf16
  float* hp[4] = {(float*)(ws + 22020096), (float*)(ws + 30408704),
                  (float*)(ws + 34603008), (float*)(ws + 35651584)};
  float* es  = (float*)(ws + 35913728);
  float* en  = (float*)(ws + 35930112);
  float* mr  = (float*)(ws + 35946496);
  float* rs  = (float*)(ws + 35962880);
  float* tmp = (float*)(ws + 35979264);                      // [512,16]

  for (int l = 0; l < 4; ++l) {
    const int K = Kin[l], d = dOut[l];
    const int mixCount = NN * K;
    if (l == 0)
      k_convert_bf16<<<(mixCount + 255) / 256, 256, 0, stream>>>(x, mixB, mixCount);
    else
      k_mix_bf16<<<(mixCount + 255) / 256, 256, 0, stream>>>(hp[l - 1], skips[l],
                                                             mixB, mixCount);
    // W^T bf16 for contiguous B fragments
    k_transpose_bf16<<<dim3((d + 31) / 32, (K + 31) / 32), dim3(32, 8), 0, stream>>>(
        W[l], WT, K, d);
    // hW = mix @ W
    k_wmma_gemm<<<dim3(d / 16, NN / 16), 32, 0, stream>>>(mixB, WT, hW, d, K);
    // attention scalars
    k_gemv2<<<NN, 32, 0, stream>>>(hW, As[l], An[l], es, en, d);
    // hW^T bf16 for attention B operand
    k_transpose_bf16<<<dim3((d + 31) / 32, (NN + 31) / 32), dim3(32, 8), 0, stream>>>(
        hW, hWT, NN, d);
    // softmax stats then fused P@h with ELU epilogue
    k_stats<<<NN, 32, 0, stream>>>(es, en, Mmat, adj, mr, rs);
    k_attn_gemm<<<dim3(NN / 16, (d + 63) / 64), 128, 0, stream>>>(
        Mmat, adj, es, en, mr, rs, hWT, hp[l], d);
  }
  // z = normalize( h1 @ (h1^T @ h4) )
  k_h1t_h4<<<512, 32, 0, stream>>>(hp[0], hp[3], tmp);
  k_z<<<NN, 32, 0, stream>>>(hp[0], tmp, (float*)d_out);
}
